// Attention_50156628082712
// MI455X (gfx1250) — matrix operand order repaired and checked by reading, not compiled, not run on hardware
//
#include <hip/hip_runtime.h>
#include <hip/hip_bf16.h>
#include <math.h>

// ---------------------------------------------------------------------------
// Shapes from the reference: b=32, t_k=2048, n=1024, all float32.
// ---------------------------------------------------------------------------
#define B_SZ  32
#define T_K   2048
#define N_DIM 1024

typedef __attribute__((ext_vector_type(16))) __bf16 v16bf;
typedef __attribute__((ext_vector_type(8)))  float  v8f;

// native f32 -> bf16 conversions, packed pair (low half = first element)
__device__ inline unsigned int cvt2_bf16(float x, float y) {
    union { __bf16 b[2]; unsigned int u; } r;
    r.b[0] = (__bf16)x;
    r.b[1] = (__bf16)y;
    return r.u;
}

// branchless tanh: 1 - 2/(exp2(2x*log2e)+1); saturates to +-1, no exec masking
__device__ inline float tanh_fast(float x) {
    float ex = __builtin_amdgcn_exp2f(x * 2.8853900817779268f); // 2*log2(e)
    float r  = __builtin_amdgcn_rcpf(ex + 1.0f);
    return 1.0f - 2.0f * r;
}

// ---------------------------------------------------------------------------
// Kernel 1: W_h (fp32 [N][N], row = output feature n', col = k)
//           -> bf16 packed pairs in workspace (same layout).
// ---------------------------------------------------------------------------
__global__ __launch_bounds__(256)
void convert_wh(const float* __restrict__ wh, unsigned int* __restrict__ out) {
    int i = blockIdx.x * 256 + threadIdx.x;           // pair index
    float2 f = *(const float2*)(wh + (size_t)i * 2);
    out[i] = cvt2_bf16(f.x, f.y);
}

// ---------------------------------------------------------------------------
// Kernel 2: dec_fea[b][n'] = dot(s_t_hat[b,:], W_dec[n',:]) + b_dec[n']
// ---------------------------------------------------------------------------
__global__ __launch_bounds__(256)
void dec_kernel(const float* __restrict__ s, const float* __restrict__ Wd,
                const float* __restrict__ bd, float* __restrict__ dec) {
    int b = blockIdx.y;
    int n = blockIdx.x * 256 + threadIdx.x;
    const float4* sr = (const float4*)(s + (size_t)b * N_DIM);
    const float4* wr = (const float4*)(Wd + (size_t)n * N_DIM);
    float acc = bd[n];
    for (int k = 0; k < N_DIM / 4; ++k) {
        float4 a = sr[k], w = wr[k];
        acc += a.x * w.x + a.y * w.y + a.z * w.z + a.w * w.w;
    }
    dec[b * N_DIM + n] = acc;
}

// ---------------------------------------------------------------------------
// Kernel 3 (main): 4-wave workgroup per (batch, 16-row t-tile).
//   scores[b][t] = sum_n' tanh(enc_feat + dec_fea + cov*w_c) * v[n']
// The 16x1024 h tile is converted to bf16 ONCE into LDS (padded rows), then
// each wave runs 2 n-groups of 8 register-blocked N-tiles, K = 1024 in steps
// of 32 via v_wmma_f32_16x16x32_bf16 (A from ds_load_b128, B from L2).
// ---------------------------------------------------------------------------
#define AROW 516   // 512 u32 (1024 bf16) + 4 u32 pad -> bank-conflict-free rows

__global__ __launch_bounds__(128)
void score_kernel(const float* __restrict__ h,
                  const unsigned int* __restrict__ whbf,   // bf16 pairs [N][N/2]
                  const float* __restrict__ dec_fea,
                  const float* __restrict__ coverage,
                  const float* __restrict__ w_c,
                  const float* __restrict__ vvec,
                  float* __restrict__ scores) {
    __shared__ unsigned int alds[16 * AROW];   // 33 KB bf16 A tile
    __shared__ float sred[4][16];              // per-wave score partials

    const int tt   = blockIdx.x;               // t-tile index, 0..127
    const int b    = blockIdx.y;               // batch
    const int tid  = threadIdx.x;
    const int w    = tid >> 5;                 // wave 0..3
    const int lane = tid & 31;
    const int m    = lane & 15;
    const int grp  = lane >> 4;                // 0 (lanes 0-15) / 1 (lanes 16-31)
    const int t0   = tt * 16;

    // ---- cooperative fp32 -> bf16 conversion of the 16x1024 h tile --------
    const float* hbase = h + ((size_t)(b * T_K + t0)) * N_DIM;
    for (int idx = tid; idx < 16 * 256; idx += 128) {
        int row = idx >> 8;                    // 256 float4 per row
        int c4  = idx & 255;
        float4 f = *(const float4*)(hbase + (size_t)row * N_DIM + c4 * 4);
        uint2 p;
        p.x = cvt2_bf16(f.x, f.y);
        p.y = cvt2_bf16(f.z, f.w);
        *(uint2*)(alds + row * AROW + c4 * 2) = p;
    }
    __syncthreads();

    // coverage per C-row: C VGPR j / this lane holds row t0 + j + 8*grp
    float covj[8];
#pragma unroll
    for (int j = 0; j < 8; ++j)
        covj[j] = coverage[b * T_K + t0 + j + grp * 8];

    float sacc[8];
#pragma unroll
    for (int j = 0; j < 8; ++j) sacc[j] = 0.0f;

    const unsigned int* abase = alds + m * AROW;

    for (int gg = 0; gg < 2; ++gg) {           // each wave: 2 of the 8 n-groups
        const int g = w * 2 + gg;
        v8f C[8];
#pragma unroll
        for (int i = 0; i < 8; ++i)
#pragma unroll
            for (int j = 0; j < 8; ++j) C[i][j] = 0.0f;

        for (int k0 = 0; k0 < N_DIM; k0 += 32) {
            // ---- A fragment from LDS (ISA 16-bit A layout):
            //   lanes 0-15:  VGPR0..3 = K k0+0..7,  VGPR4..7 = K k0+16..23
            //   lanes 16-31: VGPR0..3 = K k0+8..15, VGPR4..7 = K k0+24..31
            union { v16bf v; uint4 q[2]; } afrag;
            const unsigned int* ap = abase + (k0 >> 1) + grp * 4;
            afrag.q[0] = *(const uint4*)(ap);
            afrag.q[1] = *(const uint4*)(ap + 8);

#pragma unroll
            for (int i = 0; i < 8; ++i) {
                // ---- B fragment: 32x16 bf16 of W_h^T. Lane holds column
                // n' = tile*16 + m, K = k0 + 8*grp .. +7 and k0 + 16 + 8*grp .. +7 -> two b128 loads.
                int ncol = (g * 8 + i) * 16 + m;
                union { v16bf v; uint4 q[2]; } bfrag;
                const uint4* bp = (const uint4*)(whbf + (size_t)ncol * (N_DIM / 2)
                                                 + (unsigned)(k0 + grp * 8) / 2u);
                bfrag.q[0] = bp[0];
                bfrag.q[1] = bp[2];
                C[i] = __builtin_amdgcn_wmma_f32_16x16x32_bf16(
                           false, afrag.v, false, bfrag.v,
                           (short)0, C[i], false, false);
            }
        }

        // ---- fused epilogue: + dec_fea + cov*w_c, tanh, dot with v --------
#pragma unroll
        for (int i = 0; i < 8; ++i) {
            int ncol  = (g * 8 + i) * 16 + m;
            float dn  = dec_fea[b * N_DIM + ncol];
            float wcn = w_c[ncol];
            float vn  = vvec[ncol];
#pragma unroll
            for (int j = 0; j < 8; ++j) {
                float e = tanh_fast(C[i][j] + dn + covj[j] * wcn);
                sacc[j] += e * vn;
            }
        }
    }

    // ---- reduce: 16 lanes per half hold the 16 columns of each row --------
#pragma unroll
    for (int j = 0; j < 8; ++j) {
        float s = sacc[j];
        for (int off = 1; off < 16; off <<= 1)
            s += __shfl_xor(s, off, 32);
        if (lane == 0)        sred[w][j]     = s;
        else if (lane == 16)  sred[w][8 + j] = s;
    }
    __syncthreads();
    if (tid < 16) {
        float s = sred[0][tid] + sred[1][tid] + sred[2][tid] + sred[3][tid];
        scores[b * T_K + t0 + tid] = s;
    }
}

// ---------------------------------------------------------------------------
// Kernel 4: masked+renormalized softmax. softmax(s)*mask renormalized ==
// exp(s-max)*mask / sum(exp(s-max)*mask)  (the softmax denominator cancels).
// Also writes coverage_new = coverage + attn.
// ---------------------------------------------------------------------------
__global__ __launch_bounds__(256)
void softmax_kernel(float* scores,                       // reused as p scratch
                    const float* __restrict__ mask,
                    const float* __restrict__ coverage,
                    float* __restrict__ attn_out,
                    float* __restrict__ cov_out) {
    __shared__ float red[256];
    const int b = blockIdx.x, tid = threadIdx.x;
    float* srow = scores + (size_t)b * T_K;

    float lmax = -3.402823466e38f;
    for (int t = tid; t < T_K; t += 256) lmax = fmaxf(lmax, srow[t]);
    red[tid] = lmax; __syncthreads();
    for (int s = 128; s > 0; s >>= 1) {
        if (tid < s) red[tid] = fmaxf(red[tid], red[tid + s]);
        __syncthreads();
    }
    float mx = red[0]; __syncthreads();

    float lsum = 0.0f;
    for (int t = tid; t < T_K; t += 256) {
        float p = __builtin_amdgcn_exp2f((srow[t] - mx) * 1.4426950408889634f)
                  * mask[b * T_K + t];
        srow[t] = p;
        lsum += p;
    }
    red[tid] = lsum; __syncthreads();
    for (int s = 128; s > 0; s >>= 1) {
        if (tid < s) red[tid] += red[tid + s];
        __syncthreads();
    }
    float inv = 1.0f / red[0];

    for (int t = tid; t < T_K; t += 256) {
        float a = srow[t] * inv;
        attn_out[b * T_K + t] = a;
        cov_out[b * T_K + t]  = coverage[b * T_K + t] + a;
    }
}

// ---------------------------------------------------------------------------
// Kernel 5: c_t[b][n'] = sum_t attn[b][t] * h[b][t][n']   (bandwidth bound)
// ---------------------------------------------------------------------------
__global__ __launch_bounds__(256)
void ctx_kernel(const float* __restrict__ h, const float* __restrict__ attn,
                float* __restrict__ ct) {
    __shared__ float ash[256];
    const int b = blockIdx.y;
    const int n = blockIdx.x * 256 + threadIdx.x;
    const float* hp = h + (size_t)b * T_K * N_DIM + n;
    const float* at = attn + (size_t)b * T_K;
    float acc = 0.0f;
    for (int tb = 0; tb < T_K; tb += 256) {
        __syncthreads();
        ash[threadIdx.x] = at[tb + threadIdx.x];
        __syncthreads();
#pragma unroll 8
        for (int i = 0; i < 256; ++i)
            acc += ash[i] * hp[(size_t)(tb + i) * N_DIM];
    }
    ct[b * N_DIM + n] = acc;
}

// ---------------------------------------------------------------------------
extern "C" void kernel_launch(void* const* d_in, const int* in_sizes, int n_in,
                              void* d_out, int out_size, void* d_ws, size_t ws_size,
                              hipStream_t stream) {
    const float* s_t_hat  = (const float*)d_in[0];   // [32,1024]
    const float* h        = (const float*)d_in[1];   // [32,2048,1024]
    const float* mask     = (const float*)d_in[2];   // [32,2048]
    const float* coverage = (const float*)d_in[3];   // [32,2048]
    const float* W_h      = (const float*)d_in[4];   // [1024,1024]
    const float* W_dec    = (const float*)d_in[5];   // [1024,1024]
    const float* b_dec    = (const float*)d_in[6];   // [1024]
    const float* w_c      = (const float*)d_in[7];   // [1024]
    const float* v        = (const float*)d_in[8];   // [1024]

    // outputs, concatenated flat in return order: c_t, attn, coverage_new
    float* out_ct   = (float*)d_out;                  // 32*1024
    float* out_attn = out_ct + B_SZ * N_DIM;          // 32*2048
    float* out_cov  = out_attn + B_SZ * T_K;          // 32*2048

    // workspace layout: bf16 W_h (2 MB) | dec_fea (128 KB) | scores (256 KB)
    unsigned int* whbf  = (unsigned int*)d_ws;
    float* dec_fea = (float*)((char*)d_ws + (size_t)N_DIM * N_DIM * 2);
    float* scores  = dec_fea + B_SZ * N_DIM;

    convert_wh<<<dim3((N_DIM * N_DIM / 2) / 256), dim3(256), 0, stream>>>(W_h, whbf);
    dec_kernel<<<dim3(N_DIM / 256, B_SZ), dim3(256), 0, stream>>>(s_t_hat, W_dec, b_dec, dec_fea);
    score_kernel<<<dim3(T_K / 16, B_SZ), dim3(128), 0, stream>>>(h, whbf, dec_fea, coverage,
                                                                 w_c, v, scores);
    softmax_kernel<<<dim3(B_SZ), dim3(256), 0, stream>>>(scores, mask, coverage,
                                                         out_attn, out_cov);
    ctx_kernel<<<dim3(N_DIM / 256, B_SZ), dim3(256), 0, stream>>>(h, out_attn, out_ct);
}
